// MultiHeadedAttention_58256936403333
// MI455X (gfx1250) — matrix operand order repaired and checked by reading, not compiled, not run on hardware
//
#include <hip/hip_runtime.h>
#include <hip/hip_bf16.h>

// ---------------------------------------------------------------------------
// Multi-headed attention forward for MI455X (gfx1250, wave32, WMMA).
// All GEMMs use v_wmma_f32_16x16x32_bf16 (bf16 in, f32 accumulate).
// Projection/output GEMMs are register-blocked 32x64 per wave (8 WMMA per
// K-step from 6 fragment loads). Scores/softmax fp32; 16x2048 score strip
// in 128KB LDS (2 workgroups / 320KB WGP).
// ---------------------------------------------------------------------------

#define BATCH   4
#define L_SEQ   2048
#define D_MODEL 1024
#define NHEAD   16
#define DHEAD   64

typedef __bf16 bf16_t;
typedef bf16_t bf16x16 __attribute__((ext_vector_type(16)));
typedef float  f32x8   __attribute__((ext_vector_type(8)));

union BF16Frag {
    bf16x16  v;
    unsigned u32[8];
    bf16_t   e[16];
};

// ---- WMMA fragment loaders (layouts per CDNA5 ISA 7.12.2, wave32) ---------

// A-matrix 16x32 bf16, source row-major with K contiguous.
// Lane 0-15: M=lane, K in {0..7,16..23}; lane 16-31: M=lane-16, {8..15,24..31}.
// Per lane this is two contiguous 16B runs -> compiler emits global_load_b128.
__device__ inline bf16x16 load_frag_a_rowmajor(const bf16_t* base, int ld) {
    int lane  = threadIdx.x & 31;
    int m     = lane & 15;
    int khalf = (lane >> 4) & 1;
    BF16Frag f;
    const bf16_t* row = base + (size_t)m * ld;
#pragma unroll
    for (int v = 0; v < 8; ++v) {
        int kb = ((v & 4) << 2) + khalf * 8 + ((v & 3) << 1); // (v<4?0:16)+khalf*8+2*(v%4)
        f.u32[v] = *(const unsigned*)(row + kb);
    }
    return f.v;
}

// Same A layout but source is fp32 (LDS attention probabilities) -> cvt to bf16.
__device__ inline bf16x16 load_frag_a_f32(const float* base, int ld) {
    int lane  = threadIdx.x & 31;
    int m     = lane & 15;
    int khalf = (lane >> 4) & 1;
    BF16Frag f;
    const float* row = base + (size_t)m * ld;
#pragma unroll
    for (int v = 0; v < 8; ++v) {
        int kb = ((v & 4) << 2) + khalf * 8 + ((v & 3) << 1);
        f.e[2 * v]     = (bf16_t)row[kb];
        f.e[2 * v + 1] = (bf16_t)row[kb + 1];
    }
    return f.v;
}

// B-matrix 32x16 bf16 from K-contiguous storage Bt[n][k] (ld = row stride in k).
// Lane 0-15: N=lane, K in {0..7,16..23}; lane 16-31: N=lane-16, {8..15,24..31}.
// Per lane: two contiguous 16B runs.
__device__ inline bf16x16 load_frag_b_kcontig(const bf16_t* base, int ld) {
    int lane  = threadIdx.x & 31;
    int n     = lane & 15;
    int khalf = (lane >> 4) & 1;
    BF16Frag f;
    const bf16_t* row = base + (size_t)n * ld + khalf * 8;
#pragma unroll
    for (int v = 0; v < 8; ++v)
        f.u32[v] = *(const unsigned*)(row + ((v & 4) << 2) + ((v & 3) << 1));
    return f.v;
}

__device__ inline f32x8 wmma_bf16(bf16x16 a, bf16x16 b, f32x8 c) {
    return __builtin_amdgcn_wmma_f32_16x16x32_bf16(false, a, false, b,
                                                   (short)0, c, false, false);
}

// ---- elementwise conversion / transpose -----------------------------------

__global__ __launch_bounds__(256) void cvt_f32_bf16(const float* __restrict__ src,
                                                    bf16_t* __restrict__ dst, int n) {
    int i = blockIdx.x * 256 + threadIdx.x;
    if (i < n) dst[i] = (bf16_t)src[i];
}

// Wt[n][k] = bf16(W[k][n]); makes B-fragment dword loads contiguous.
__global__ __launch_bounds__(256) void transpose_w(const float* __restrict__ W,
                                                   bf16_t* __restrict__ Wt) {
    int i = blockIdx.x * 256 + threadIdx.x;
    if (i < D_MODEL * D_MODEL) {
        int n = i / D_MODEL, k = i - n * D_MODEL;
        Wt[i] = (bf16_t)W[(size_t)k * D_MODEL + n];
    }
}

// ---- fused Q/K/V projection GEMM (register-blocked) -----------------------
// blockIdx.z selects {0:q,1:k,2:v}. One wave computes a 32x64 output tile:
// per K-step, 2 A-frags x 4 B-frags -> 8 WMMA (0.75 fragment loads / WMMA).
// Block = 8 waves as 4(m) x 2(n) -> 128x128 block tile.
// q_up/k_up -> (B,H,L,DH); v -> transposed (B,H,DH,L) for K-contiguous PV.
__global__ __launch_bounds__(256) void proj_gemm(
    const bf16_t* __restrict__ qx, const bf16_t* __restrict__ kx,
    const bf16_t* __restrict__ vx,
    const bf16_t* __restrict__ WqT, const bf16_t* __restrict__ WkT,
    const bf16_t* __restrict__ WvT,
    const float* __restrict__ bq, const float* __restrict__ bk,
    const float* __restrict__ bv,
    bf16_t* __restrict__ q_up, bf16_t* __restrict__ k_up,
    bf16_t* __restrict__ v_upT) {
    int which = blockIdx.z;
    const bf16_t* X    = (which == 0) ? qx  : (which == 1) ? kx  : vx;
    const bf16_t* Wt   = (which == 0) ? WqT : (which == 1) ? WkT : WvT;
    const float*  bias = (which == 0) ? bq  : (which == 1) ? bk  : bv;

    int wave = threadIdx.x >> 5;
    int m0 = blockIdx.y * 128 + (wave >> 1) * 32;   // 2 m-tiles of 16
    int n0 = blockIdx.x * 128 + (wave & 1) * 64;    // 4 n-tiles of 16

    const bf16_t* abase = X  + (size_t)m0 * D_MODEL;
    const bf16_t* bbase = Wt + (size_t)n0 * D_MODEL;

    f32x8 acc[2][4] = {{{}, {}, {}, {}}, {{}, {}, {}, {}}};
    for (int k0 = 0; k0 < D_MODEL; k0 += 32) {
        bf16x16 a[2], b[4];
#pragma unroll
        for (int i = 0; i < 2; ++i)
            a[i] = load_frag_a_rowmajor(abase + (size_t)(i * 16) * D_MODEL + k0,
                                        D_MODEL);
#pragma unroll
        for (int t = 0; t < 4; ++t)
            b[t] = load_frag_b_kcontig(bbase + (size_t)(t * 16) * D_MODEL + k0,
                                       D_MODEL);
#pragma unroll
        for (int i = 0; i < 2; ++i)
#pragma unroll
            for (int t = 0; t < 4; ++t)
                acc[i][t] = wmma_bf16(a[i], b[t], acc[i][t]);
    }

    int lane  = threadIdx.x & 31;
    float scale = (which == 0) ? 0.125f : 1.0f;  // q scaled by 1/sqrt(DH)
#pragma unroll
    for (int t = 0; t < 4; ++t) {
        int ng = n0 + t * 16 + (lane & 15);
        int h  = ng >> 6, dh = ng & 63;
        float bval = bias[ng];
#pragma unroll
        for (int i = 0; i < 2; ++i)
#pragma unroll
            for (int r = 0; r < 8; ++r) {
                int   mg  = m0 + i * 16 + r + ((lane >> 4) << 3);
                int   bb  = mg >> 11, l = mg & (L_SEQ - 1);
                float val = (acc[i][t][r] + bval) * scale;
                bf16_t o  = (bf16_t)val;
                if (which == 2)
                    v_upT[((size_t)(bb * NHEAD + h) * DHEAD + dh) * L_SEQ + l] = o;
                else if (which == 0)
                    q_up[((size_t)(bb * NHEAD + h) * L_SEQ + l) * DHEAD + dh] = o;
                else
                    k_up[((size_t)(bb * NHEAD + h) * L_SEQ + l) * DHEAD + dh] = o;
            }
    }
}

// ---- attention core --------------------------------------------------------
// One workgroup = (b, h, 16-query block). 16 x L fp32 score strip in LDS.
__global__ __launch_bounds__(256) void attn_kernel(
    const bf16_t* __restrict__ q_up, const bf16_t* __restrict__ k_up,
    const bf16_t* __restrict__ v_upT, bf16_t* __restrict__ ctx,
    float* __restrict__ top_attn) {
    __shared__ float sc[16 * L_SEQ];  // 128 KB -> 2 workgroups per 320KB WGP

    int qblk = blockIdx.x;
    int h    = blockIdx.y;
    int b    = blockIdx.z;
    int q0   = qblk * 16;
    int nlim = q0 + 16;          // causal column bound for this strip
    int wave = threadIdx.x >> 5;
    int lane = threadIdx.x & 31;

    const bf16_t* qbase = q_up  + ((size_t)(b * NHEAD + h) * L_SEQ + q0) * DHEAD;
    const bf16_t* kbase = k_up  +  (size_t)(b * NHEAD + h) * L_SEQ * DHEAD;
    const bf16_t* vbase = v_upT +  (size_t)(b * NHEAD + h) * DHEAD * L_SEQ;

    // Q fragments (DH=64 -> two K=32 chunks), reused for every key tile.
    bf16x16 qa0 = load_frag_a_rowmajor(qbase + 0,  DHEAD);
    bf16x16 qa1 = load_frag_a_rowmajor(qbase + 32, DHEAD);

    // --- scores = (q/sqrt(dh)) . k^T, causal-masked, striped over waves ---
    int nktiles = qblk + 1;
    for (int kt = wave; kt < nktiles; kt += 8) {
        const bf16_t* kb = kbase + (size_t)(kt * 16) * DHEAD;
        f32x8 s = {};
        s = wmma_bf16(qa0, load_frag_b_kcontig(kb + 0,  DHEAD), s);
        s = wmma_bf16(qa1, load_frag_b_kcontig(kb + 32, DHEAD), s);
        int ncol = kt * 16 + (lane & 15);
#pragma unroll
        for (int r = 0; r < 8; ++r) {
            int m  = r + ((lane >> 4) << 3);
            int qg = q0 + m;
            sc[m * L_SEQ + ncol] = (ncol > qg) ? -1e30f : s[r];
        }
    }
    __syncthreads();

    // --- fp32 softmax: wave w owns rows 2w, 2w+1 (wave32 shuffle reduce) ---
    int n32 = (nlim + 31) & ~31;
#pragma unroll
    for (int rr = 0; rr < 2; ++rr) {
        int m = wave * 2 + rr;
        float mx = -3.0e38f;
        for (int c = lane; c < nlim; c += 32) mx = fmaxf(mx, sc[m * L_SEQ + c]);
        for (int off = 16; off; off >>= 1) mx = fmaxf(mx, __shfl_xor(mx, off, 32));
        float sum = 0.f;
        for (int c = lane; c < nlim; c += 32) sum += __expf(sc[m * L_SEQ + c] - mx);
        for (int off = 16; off; off >>= 1) sum += __shfl_xor(sum, off, 32);
        float inv = 1.0f / sum;
        for (int c = lane; c < n32; c += 32) {
            float v = (c < nlim) ? __expf(sc[m * L_SEQ + c] - mx) * inv : 0.f;
            sc[m * L_SEQ + c] = v;  // zero-pad to 32-alignment for WMMA chunks
        }
    }
    __syncthreads();

    // --- head-0 attention probabilities out to d_out ---
    if (h == 0) {
        for (int idx = threadIdx.x; idx < 16 * L_SEQ; idx += 256) {
            int m = idx >> 11, c = idx & (L_SEQ - 1);
            float v = (c < nlim) ? sc[m * L_SEQ + c] : 0.f;
            top_attn[((size_t)b * L_SEQ + (q0 + m)) * L_SEQ + c] = v;
        }
    }

    // --- ctx = attn @ V, K-parallel across waves, then LDS reduce ---
    int nchunks = n32 >> 5;
    f32x8 accs[4] = {{}, {}, {}, {}};
    for (int c = wave; c < nchunks; c += 8) {
        bf16x16 a = load_frag_a_f32(sc + c * 32, L_SEQ);  // reused for 4 B-frags
#pragma unroll
        for (int t = 0; t < 4; ++t) {
            bf16x16 bb = load_frag_b_kcontig(vbase + (size_t)(t * 16) * L_SEQ + c * 32,
                                             L_SEQ);
            accs[t] = wmma_bf16(a, bb, accs[t]);
        }
    }
    __syncthreads();  // all waves done reading sc; safe to reuse as reduce buffer

    {
        int nloc = lane & 15;
#pragma unroll
        for (int t = 0; t < 4; ++t)
#pragma unroll
            for (int r = 0; r < 8; ++r) {
                int m  = r + ((lane >> 4) << 3);
                int dh = t * 16 + nloc;
                sc[(wave * 16 + m) * DHEAD + dh] = accs[t][r];  // 32 KB reuse
            }
    }
    __syncthreads();

    for (int e = threadIdx.x; e < 16 * DHEAD; e += 256) {
        int m = e >> 6, dh = e & 63;
        float s = 0.f;
#pragma unroll
        for (int w = 0; w < 8; ++w) s += sc[(w * 16 + m) * DHEAD + dh];
        ctx[((size_t)(b * L_SEQ) + q0 + m) * D_MODEL + h * DHEAD + dh] = (bf16_t)s;
    }
}

// ---- output projection: out = ctx @ Wo + bo (fp32 out, blocked 32x64) -----
__global__ __launch_bounds__(256) void out_gemm(const bf16_t* __restrict__ ctx,
                                                const bf16_t* __restrict__ WoT,
                                                const float* __restrict__ bo,
                                                float* __restrict__ out) {
    int wave = threadIdx.x >> 5;
    int m0 = blockIdx.y * 128 + (wave >> 1) * 32;
    int n0 = blockIdx.x * 128 + (wave & 1) * 64;

    const bf16_t* abase = ctx + (size_t)m0 * D_MODEL;
    const bf16_t* bbase = WoT + (size_t)n0 * D_MODEL;

    f32x8 acc[2][4] = {{{}, {}, {}, {}}, {{}, {}, {}, {}}};
    for (int k0 = 0; k0 < D_MODEL; k0 += 32) {
        bf16x16 a[2], b[4];
#pragma unroll
        for (int i = 0; i < 2; ++i)
            a[i] = load_frag_a_rowmajor(abase + (size_t)(i * 16) * D_MODEL + k0,
                                        D_MODEL);
#pragma unroll
        for (int t = 0; t < 4; ++t)
            b[t] = load_frag_b_kcontig(bbase + (size_t)(t * 16) * D_MODEL + k0,
                                       D_MODEL);
#pragma unroll
        for (int i = 0; i < 2; ++i)
#pragma unroll
            for (int t = 0; t < 4; ++t)
                acc[i][t] = wmma_bf16(a[i], b[t], acc[i][t]);
    }

    int lane = threadIdx.x & 31;
#pragma unroll
    for (int t = 0; t < 4; ++t) {
        int ng = n0 + t * 16 + (lane & 15);
        float bval = bo[ng];
#pragma unroll
        for (int i = 0; i < 2; ++i)
#pragma unroll
            for (int r = 0; r < 8; ++r) {
                int mg = m0 + i * 16 + r + ((lane >> 4) << 3);
                out[(size_t)mg * D_MODEL + ng] = acc[i][t][r] + bval;
            }
    }
}

// ---------------------------------------------------------------------------

extern "C" void kernel_launch(void* const* d_in, const int* in_sizes, int n_in,
                              void* d_out, int out_size, void* d_ws, size_t ws_size,
                              hipStream_t stream) {
    // input order: key, value, query, mask, Wk, bk, Wv, bv, Wq, bq, Wo, bo
    const float* key_f   = (const float*)d_in[0];
    const float* value_f = (const float*)d_in[1];
    const float* query_f = (const float*)d_in[2];
    const float* Wk = (const float*)d_in[4];
    const float* bk = (const float*)d_in[5];
    const float* Wv = (const float*)d_in[6];
    const float* bv = (const float*)d_in[7];
    const float* Wq = (const float*)d_in[8];
    const float* bq = (const float*)d_in[9];
    const float* Wo = (const float*)d_in[10];
    const float* bo = (const float*)d_in[11];

    const size_t nBLD = (size_t)BATCH * L_SEQ * D_MODEL;  // 8388608
    const size_t nDD  = (size_t)D_MODEL * D_MODEL;        // 1048576

    char* ws = (char*)d_ws;
    size_t off = 0;
    auto take = [&](size_t bytes) {
        size_t o = off;
        off = (off + bytes + 255) & ~(size_t)255;
        return o;
    };
    bf16_t* qbf   = (bf16_t*)(ws + take(nBLD * 2));
    bf16_t* kbf   = (bf16_t*)(ws + take(nBLD * 2));
    bf16_t* vbf   = (bf16_t*)(ws + take(nBLD * 2));
    bf16_t* WqT   = (bf16_t*)(ws + take(nDD * 2));
    bf16_t* WkT   = (bf16_t*)(ws + take(nDD * 2));
    bf16_t* WvT   = (bf16_t*)(ws + take(nDD * 2));
    bf16_t* WoT   = (bf16_t*)(ws + take(nDD * 2));
    bf16_t* q_up  = (bf16_t*)(ws + take(nBLD * 2));
    bf16_t* k_up  = (bf16_t*)(ws + take(nBLD * 2));
    bf16_t* v_upT = (bf16_t*)(ws + take(nBLD * 2));
    bf16_t* ctx   = (bf16_t*)(ws + take(nBLD * 2));

    float* out_f    = (float*)d_out;                 // (B,L,D)
    float* top_attn = out_f + nBLD;                  // (B,L,L)

    // 1) casts + weight transposes
    {
        int n = (int)nBLD;
        dim3 g((n + 255) / 256);
        cvt_f32_bf16<<<g, 256, 0, stream>>>(query_f, qbf, n);
        cvt_f32_bf16<<<g, 256, 0, stream>>>(key_f,   kbf, n);
        cvt_f32_bf16<<<g, 256, 0, stream>>>(value_f, vbf, n);
        dim3 gw((int)(nDD + 255) / 256);
        transpose_w<<<gw, 256, 0, stream>>>(Wq, WqT);
        transpose_w<<<gw, 256, 0, stream>>>(Wk, WkT);
        transpose_w<<<gw, 256, 0, stream>>>(Wv, WvT);
        transpose_w<<<gw, 256, 0, stream>>>(Wo, WoT);
    }

    // 2) fused Q/K/V projections (WMMA, 32x64 per wave)
    {
        dim3 g(D_MODEL / 128, (BATCH * L_SEQ) / 128, 3);  // 8 x 64 x 3
        proj_gemm<<<g, 256, 0, stream>>>(qbf, kbf, vbf, WqT, WkT, WvT,
                                         bq, bk, bv, q_up, k_up, v_upT);
    }

    // 3) attention core (WMMA + LDS softmax)
    {
        dim3 g(L_SEQ / 16, NHEAD, BATCH);  // 128 x 16 x 4
        attn_kernel<<<g, 256, 0, stream>>>(q_up, k_up, v_upT, ctx, top_attn);
    }

    // 4) output projection (WMMA, 32x64 per wave)
    {
        dim3 g(D_MODEL / 128, (BATCH * L_SEQ) / 128, 1);  // 8 x 64
        out_gemm<<<g, 256, 0, stream>>>(ctx, WoT, bo, out_f);
    }
}
